// AttnModel_75118978007374
// MI455X (gfx1250) — matrix operand order repaired and checked by reading, not compiled, not run on hardware
//
#include <hip/hip_runtime.h>
#include <hip/hip_bf16.h>

typedef _Float16 h16;
typedef __attribute__((ext_vector_type(16))) _Float16 v16h;
typedef __attribute__((ext_vector_type(8)))  float    v8f;

#define Bsz 8192
#define Nn  64
#define Dd  128
#define Hh  2
#define DMm 384
#define DKk 192
#define GB  16      // batches per workgroup
#define KSF 388     // kin LDS stride (floats): 1552B = 97*16 (16B aligned), 4-bank row skew
#define AS  392     // actA stride (halves):    784B  = 49*16
#define BS  776     // actB / ubuf stride (halves): 1552B = 97*16, 4-bank row skew
#define OS  388     // outf stride (floats)

// packed-weight offsets in halves within d_ws
#define OFF_WQ  0
#define OFF_WKT 147456
#define OFF_WV  294912
#define OFF_WFC 442368
#define OFF_W1  589824
#define OFF_W2  655360

__device__ __forceinline__ v8f wmma16(v16h a, v16h b, v8f c) {
  // D = A(16x32 f16) * B(32x16 f16) + C(16x16 f32)
  return __builtin_amdgcn_wmma_f32_16x16x32_f16(false, a, false, b, (short)0, c, false, false);
}

// Generic (AS0) LDS pointer -> 32-bit LDS byte offset for async-to-LDS VDST.
__device__ __forceinline__ unsigned lds_off_of(const void* p) {
  return (unsigned)(uintptr_t)(const __attribute__((address_space(3))) void*)p;
}

// A fragment (16x32, MxK) from row-major f16 LDS. Per ISA layout:
// lane: m=lane&15, koff=(lane>>4)*8; halves j<8 -> K=koff+j, j>=8 -> K=16+koff+(j-8)
__device__ __forceinline__ v16h load_a_lds(const h16* base, int stride, int k0) {
  int lane = threadIdx.x & 31;
  int m    = lane & 15;
  int koff = (lane >> 4) << 3;
  const h16* p = base + m * stride + k0 + koff;
  v16h a;
#pragma unroll
  for (int j = 0; j < 8; ++j) a[j]     = p[j];
#pragma unroll
  for (int j = 0; j < 8; ++j) a[j + 8] = p[16 + j];
  return a;
}

// B fragment (32x16, KxN) pre-packed in global: frag*512 + lane*16 + j
__device__ __forceinline__ v16h load_b_pack(const h16* pk, int frag) {
  int lane = threadIdx.x & 31;
  return *(const v16h*)(pk + ((size_t)frag << 9) + (lane << 4));
}

// C fragment store to f16 LDS: m = r + 8*(lane>>4), n = n0 + (lane&15)
__device__ __forceinline__ void store_c_f16(h16* base, int stride, int n0, v8f c) {
  int lane = threadIdx.x & 31;
  int n  = n0 + (lane & 15);
  int mb = (lane >> 4) << 3;
#pragma unroll
  for (int r = 0; r < 8; ++r) base[(mb + r) * stride + n] = (h16)c[r];
}

// Pack a KxN B-operand (f32 source) into WMMA B-fragment order (f16).
// transpose==0: B[k,n] = W[(row_off+k)*ldw + col_off+n]
// transpose==1: B[k,n] = W[(col_off+n)*ldw + row_off+k]
__global__ void pack_b_kernel(const float* __restrict__ W, h16* __restrict__ out,
                              int K, int Ncols, int ldw, int row_off, int col_off,
                              int transpose) {
  int idx = blockIdx.x * 256 + threadIdx.x;
  int total = K * Ncols;
  if (idx >= total) return;
  int frag = idx >> 9, within = idx & 511;
  int lane = within >> 4, j = within & 15;
  int NT = Ncols >> 4;
  int kt = frag / NT, nt = frag - kt * NT;
  int n  = lane & 15;
  int k  = (j < 8) ? ((lane >> 4) << 3) + j : 16 + ((lane >> 4) << 3) + (j - 8);
  int kk = kt * 32 + k, nn = nt * 16 + n;
  float v = transpose ? W[(size_t)(col_off + nn) * ldw + row_off + kk]
                      : W[(size_t)(row_off + kk) * ldw + col_off + nn];
  out[idx] = (h16)v;
}

__global__ __launch_bounds__(256)
void attn_fused_kernel(const float* __restrict__ src,  const float* __restrict__ src_t,
                       const float* __restrict__ seq,  const float* __restrict__ seq_t,
                       const float* __restrict__ seq_e, const int* __restrict__ maskp,
                       const float* __restrict__ bfc,  const float* __restrict__ ln_g,
                       const float* __restrict__ ln_b, const float* __restrict__ b1,
                       const float* __restrict__ b2,   const h16* __restrict__ packs,
                       float* __restrict__ out_merged, float* __restrict__ out_attn) {
  // kin (fp32, async-DMA target) used only in the per-batch loop;
  // outf used only after it -> overlap in a union.
  __shared__ union { float kinf[Nn * KSF]; float outf[GB * OS]; } ov;
  __shared__ h16   actA[GB * AS];   // qh, later ctx, later h1
  __shared__ h16   actB[GB * BS];   // Q, later w (16 x 768), later hcat (16 x 512)
  __shared__ h16   ubuf[GB * BS];   // u = attn @ k_in, (16 x 768)
  __shared__ float scb[Hh * Nn];
  __shared__ float attnb[Hh * Nn];
  __shared__ float red[4];

  const h16* pWq  = packs + OFF_WQ;
  const h16* pWkT = packs + OFF_WKT;   // per head: 192x384, +73728/head
  const h16* pWv  = packs + OFF_WV;    // per head: 384x192, +73728/head
  const h16* pWfc = packs + OFF_WFC;
  const h16* pW1  = packs + OFF_W1;
  const h16* pW2  = packs + OFF_W2;

  const int tid  = threadIdx.x;
  const int wave = tid >> 5;
  const int lane = tid & 31;
  const int b0   = blockIdx.x * GB;

  // ---- Phase A: build Q = q_in f16 (16 x 384) in actB ----
  for (int idx = tid; idx < GB * DMm; idx += 256) {
    int i = idx / DMm, c = idx - i * DMm;
    float v;
    if (c < Dd)            v = src  [(size_t)(b0 + i) * Dd + c];
    else if (c < 2 * Dd)   v = 0.f;
    else                   v = src_t[(size_t)(b0 + i) * Dd + (c - 2 * Dd)];
    actB[i * BS + c] = (h16)v;
  }
  __syncthreads();

  // ---- Phase B: qh = Q @ Wq  (16x384) -> actA f16 ----
  for (int t = 0; t < 3; ++t) {
    int nt = wave * 3 + t;                       // 24 N-tiles over 8 waves
    v8f acc = {};
    for (int kt = 0; kt < 12; ++kt) {
      v16h a = load_a_lds(actB, BS, kt * 32);
      v16h b = load_b_pack(pWq, kt * 24 + nt);
      acc = wmma16(a, b, acc);
    }
    store_c_f16(actA, AS, nt * 16, acc);
  }
  __syncthreads();

  // ---- Phase C: w_h = qh_h @ Wk_h^T (16 x 384 per head) -> actB ----
  for (int q = 0; q < 6; ++q) {
    int id = wave * 6 + q;                       // 2 heads * 24 tiles
    int h = id / 24, nt = id - h * 24;
    v8f acc = {};
    for (int kt = 0; kt < 6; ++kt) {             // K = DK = 192
      v16h a = load_a_lds(actA, AS, h * DKk + kt * 32);
      v16h b = load_b_pack(pWkT + h * 73728, kt * 24 + nt);
      acc = wmma16(a, b, acc);
    }
    store_c_f16(actB, BS, h * DMm + nt * 16, acc);
  }
  __syncthreads();

  // ---- Phase D: stream each batch's k_in via async cache->LDS DMA;
  //               scores -> softmax -> u ----
  for (int i = 0; i < GB; ++i) {
    const int b = b0 + i;
    const size_t robase = (size_t)b * Nn * Dd;
    // k_in = [seq | seq_e | seq_t] -> kinf fp32 (64 x 384), no VGPR round-trip:
    // GLOBAL_LOAD_ASYNC_TO_LDS_B128, tracked by ASYNCcnt.
    for (int idx = tid * 4; idx < Nn * DMm; idx += 256 * 4) {
      int n = idx / DMm, c = idx - n * DMm;
      const float* sp; int cc;
      if (c < Dd)          { sp = seq;   cc = c; }
      else if (c < 2 * Dd) { sp = seq_e; cc = c - Dd; }
      else                 { sp = seq_t; cc = c - 2 * Dd; }
      const float* gaddr = sp + robase + (size_t)n * Dd + cc;
      unsigned     laddr = lds_off_of(&ov.kinf[n * KSF + c]);
      asm volatile("global_load_async_to_lds_b128 %0, %1, off"
                   :: "v"(laddr), "v"(gaddr) : "memory");
    }
    // prefetch next batch's tiles into cache while we compute this one
    if (i + 1 < GB) {
      size_t nb = (size_t)(b + 1) * Nn * Dd + (size_t)tid * 32;
      __builtin_prefetch(seq   + nb, 0, 1);
      __builtin_prefetch(seq_e + nb, 0, 1);
      __builtin_prefetch(seq_t + nb, 0, 1);
    }
    asm volatile("s_wait_asynccnt 0x0" ::: "memory");
    __syncthreads();

    // scores[h][n] = (k_in[n,:] . w_h) / sqrt(DK), masked
    if (tid < 128) {
      int h = tid >> 6, n = tid & 63;
      const h16*   wrow = &actB[i * BS + h * DMm];
      const float* krow = &ov.kinf[n * KSF];
      float s = 0.f;
      for (int c = 0; c < DMm; ++c) s += krow[c] * (float)wrow[c];
      s *= 0.07216878364870323f;                 // 1/sqrt(192)
      if (maskp[(size_t)b * Nn + n] != 0) s = -1e10f;
      scb[tid] = s;
    }
    __syncthreads();
    if (tid < 2) {                               // tiny serial row reduction
      float mx = -3.4e38f;
      for (int n = 0; n < Nn; ++n) mx = fmaxf(mx, scb[tid * Nn + n]);
      float sm = 0.f;
      for (int n = 0; n < Nn; ++n) sm += __expf(scb[tid * Nn + n] - mx);
      red[tid * 2] = mx; red[tid * 2 + 1] = sm;
    }
    __syncthreads();
    if (tid < 128) {
      int h = tid >> 6, n = tid & 63;
      float a = __expf(scb[tid] - red[h * 2]) / red[h * 2 + 1];
      attnb[tid] = a;
      out_attn[((size_t)h * Bsz + b) * Nn + n] = a;   // attn_out (H*B, N)
    }
    __syncthreads();
    // u[h][c] = sum_n attn[h][n] * k_in[n][c]  -> ubuf row i (f16)
    for (int oid = tid; oid < Hh * DMm; oid += 256) {
      int h = oid / DMm, c = oid - h * DMm;
      const float* ap = &attnb[h * Nn];
      float acc = 0.f;
      for (int n = 0; n < Nn; ++n) acc += ap[n] * ov.kinf[n * KSF + c];
      ubuf[i * BS + oid] = (h16)acc;
    }
    __syncthreads();
  }

  // ---- Phase E: ctx_h = u_h @ Wv_h  (16 x 192 per head) -> actA ----
  for (int t = 0; t < 3; ++t) {
    int id = wave * 3 + t;                       // 2 heads * 12 tiles
    int h = id / 12, nt = id - h * 12;
    v8f acc = {};
    for (int kt = 0; kt < 12; ++kt) {
      v16h a = load_a_lds(ubuf, BS, h * DMm + kt * 32);
      v16h b = load_b_pack(pWv + h * 73728, kt * 12 + nt);
      acc = wmma16(a, b, acc);
    }
    store_c_f16(actA, AS, h * DKk + nt * 16, acc);
  }
  __syncthreads();

  // ---- Phase F: out = ctx @ Wfc + bfc + q_in -> outf f32 ----
  for (int t = 0; t < 3; ++t) {
    int nt = wave * 3 + t;
    v8f acc = {};
    for (int kt = 0; kt < 12; ++kt) {
      v16h a = load_a_lds(actA, AS, kt * 32);
      v16h b = load_b_pack(pWfc, kt * 24 + nt);
      acc = wmma16(a, b, acc);
    }
    int n  = nt * 16 + (lane & 15);
    int mb = (lane >> 4) << 3;
    float bias = bfc[n];
#pragma unroll
    for (int r = 0; r < 8; ++r) {
      int m = mb + r;
      float qv;
      if (n < Dd)          qv = src  [(size_t)(b0 + m) * Dd + n];
      else if (n < 2 * Dd) qv = 0.f;
      else                 qv = src_t[(size_t)(b0 + m) * Dd + (n - 2 * Dd)];
      ov.outf[m * OS + n] = acc[r] + bias + qv;
    }
  }
  __syncthreads();

  // ---- Phase G: LayerNorm rows + build hcat=[out, src] f16 in actB ----
  for (int rr = 0; rr < 2; ++rr) {
    int m = wave * 2 + rr;
    float sum = 0.f, sq = 0.f;
    for (int c = lane; c < DMm; c += 32) { float x = ov.outf[m * OS + c]; sum += x; sq += x * x; }
#pragma unroll
    for (int off = 16; off > 0; off >>= 1) {
      sum += __shfl_xor(sum, off, 32);
      sq  += __shfl_xor(sq,  off, 32);
    }
    float mu   = sum * (1.f / DMm);
    float var  = sq * (1.f / DMm) - mu * mu;
    float rstd = rsqrtf(var + 1e-5f);
    for (int c = lane; c < DMm; c += 32) {
      float x = (ov.outf[m * OS + c] - mu) * rstd * ln_g[c] + ln_b[c];
      actB[m * BS + c] = (h16)x;
    }
  }
  for (int idx = tid; idx < GB * Dd; idx += 256) {
    int i = idx / Dd, c = idx - i * Dd;
    actB[i * BS + DMm + c] = (h16)src[(size_t)(b0 + i) * Dd + c];
  }
  __syncthreads();

  // ---- Phase H: h1 = relu(hcat @ W1 + b1) (16 x 128) -> actA ----
  {
    int nt = wave;                               // 8 tiles, one per wave
    v8f acc = {};
    for (int kt = 0; kt < 16; ++kt) {            // K = 512
      v16h a = load_a_lds(actB, BS, kt * 32);
      v16h b = load_b_pack(pW1, kt * 8 + nt);
      acc = wmma16(a, b, acc);
    }
    int n  = nt * 16 + (lane & 15);
    int mb = (lane >> 4) << 3;
    float bias = b1[n];
#pragma unroll
    for (int r = 0; r < 8; ++r)
      actA[(mb + r) * AS + n] = (h16)fmaxf(acc[r] + bias, 0.f);
  }
  __syncthreads();

  // ---- Phase I: merged = h1 @ W2 + b2 -> global ----
  {
    int nt = wave;
    v8f acc = {};
    for (int kt = 0; kt < 4; ++kt) {             // K = 128
      v16h a = load_a_lds(actA, AS, kt * 32);
      v16h b = load_b_pack(pW2, kt * 8 + nt);
      acc = wmma16(a, b, acc);
    }
    int n  = nt * 16 + (lane & 15);
    int mb = (lane >> 4) << 3;
    float bias = b2[n];
#pragma unroll
    for (int r = 0; r < 8; ++r)
      out_merged[(size_t)(b0 + mb + r) * Dd + n] = acc[r] + bias;
  }
}

extern "C" void kernel_launch(void* const* d_in, const int* in_sizes, int n_in,
                              void* d_out, int out_size, void* d_ws, size_t ws_size,
                              hipStream_t stream) {
  const float* src   = (const float*)d_in[0];
  const float* src_t = (const float*)d_in[1];
  const float* seq   = (const float*)d_in[2];
  const float* seq_t = (const float*)d_in[3];
  const float* seq_e = (const float*)d_in[4];
  const int*   maskp = (const int*)d_in[5];
  const float* Wq    = (const float*)d_in[6];
  const float* Wk    = (const float*)d_in[7];
  const float* Wv    = (const float*)d_in[8];
  const float* Wfc   = (const float*)d_in[9];
  const float* bfc   = (const float*)d_in[10];
  const float* ln_g  = (const float*)d_in[11];
  const float* ln_b  = (const float*)d_in[12];
  const float* W1    = (const float*)d_in[13];
  const float* b1    = (const float*)d_in[14];
  const float* W2    = (const float*)d_in[15];
  const float* b2    = (const float*)d_in[16];

  h16* packs = (h16*)d_ws;

  auto pk = [&](const float* W, int off, int K, int Ncols, int ldw, int ro, int co, int tr) {
    int total = K * Ncols;
    pack_b_kernel<<<(total + 255) / 256, 256, 0, stream>>>(W, packs + off, K, Ncols, ldw, ro, co, tr);
  };
  pk(Wq,  OFF_WQ,            384, 384, 384, 0,   0,   0);
  pk(Wk,  OFF_WKT,           192, 384, 384, 0,   0,   1);  // head0: B[k,n]=Wk[n, k]
  pk(Wk,  OFF_WKT + 73728,   192, 384, 384, 192, 0,   1);  // head1: B[k,n]=Wk[n, 192+k]
  pk(Wv,  OFF_WV,            384, 192, 384, 0,   0,   0);  // head0: Wv[:, 0:192]
  pk(Wv,  OFF_WV + 73728,    384, 192, 384, 0,   192, 0);  // head1: Wv[:, 192:384]
  pk(Wfc, OFF_WFC,           384, 384, 384, 0,   0,   0);
  pk(W1,  OFF_W1,            512, 128, 128, 0,   0,   0);
  pk(W2,  OFF_W2,            128, 128, 128, 0,   0,   0);

  float* out_merged = (float*)d_out;
  float* out_attn   = (float*)d_out + (size_t)Bsz * Dd;

  attn_fused_kernel<<<Bsz / GB, 256, 0, stream>>>(
      src, src_t, seq, seq_t, seq_e, maskp, bfc, ln_g, ln_b, b1, b2,
      packs, out_merged, out_attn);
}
